// TokenizerEncoder_4449586118753
// MI455X (gfx1250) — matrix operand order repaired and checked by reading, not compiled, not run on hardware
//
#include <hip/hip_runtime.h>
#include <hip/hip_bf16.h>
#include <math.h>

// ---------------------------------------------------------------------------
// Types for WMMA (gfx1250, wave32)
// ---------------------------------------------------------------------------
typedef __attribute__((ext_vector_type(16))) _Float16 v16h;
typedef __attribute__((ext_vector_type(8)))  _Float16 v8h;
typedef __attribute__((ext_vector_type(8)))  float    v8f;

// Model constants
#define BB 2
#define TT 8
#define NTOK 576
#define NLAT 32
#define TOTS 608          // NTOK + NLAT
#define DM 512
#define NH 8
#define NKV 4
#define HD 64
#define MLPD 2048
#define CIN 768
#define MROWS (BB*TT*TOTS)   // 9728
#define MTOK  (BB*TT*NTOK)   // 9216

// ---------------------------------------------------------------------------
// CDNA5 async copy: global -> LDS, 16 bytes per lane, tracked by ASYNCcnt.
// (ISA 15.18.3 opcode 98 GLOBAL_LOAD_ASYNC_TO_LDS_B128, GV addressing mode.)
// ---------------------------------------------------------------------------
__device__ inline void async_copy_b128(unsigned lds_off, const void* gaddr) {
  asm volatile("global_load_async_to_lds_b128 %0, %1, off"
               :: "v"(lds_off), "v"(gaddr) : "memory");
}
__device__ inline void wait_async_le1() {
  asm volatile("s_wait_asynccnt 0x1" ::: "memory");
}
__device__ inline void wait_async_0() {
  asm volatile("s_wait_asynccnt 0x0" ::: "memory");
}

// ---------------------------------------------------------------------------
// WMMA fragment loaders.
//
// A (16-bit, 16x32 MxK), row-major with stride `ld` (from global):
//   lanes 0-15  : row M=lane,    elems 0..15 = K k0..k0+15
//   lanes 16-31 : row M=lane-16, elems 0..15 = K k0+16..k0+31
// B (16-bit, 32x16 KxN), stored K-major (Bt[n][k]):
//   lanes 0-15  : col N=lane,    elems 0..15 = K k0..k0+15
//   lanes 16-31 : col N=lane-16, elems 0..15 = K k0+16..k0+31
// ---------------------------------------------------------------------------
__device__ inline v16h load_frag_a(const _Float16* __restrict__ A, int ld,
                                   int row0, int k0, int lane) {
  const int l = lane & 15;
  const int hi = (lane >= 16) ? 16 : 0;
  const _Float16* p = A + (size_t)(row0 + l) * ld + k0 + hi;
  v8h lo = *(const v8h*)p;
  v8h hh = *(const v8h*)(p + 8);
  return __builtin_shufflevector(lo, hh, 0,1,2,3,4,5,6,7,8,9,10,11,12,13,14,15);
}

// B fragment from global (used by the one-shot input-projection GEMM).
__device__ inline v16h load_frag_b(const _Float16* __restrict__ Bt, int ld,
                                   int col0, int k0, int lane) {
  const int l = lane & 15;
  const int hi = (lane >= 16) ? 16 : 0;
  const _Float16* p = Bt + (size_t)(col0 + l) * ld + k0 + hi;
  v8h lo = *(const v8h*)p;
  v8h hh = *(const v8h*)(p + 8);
  return __builtin_shufflevector(lo, hh, 0,1,2,3,4,5,6,7,8,9,10,11,12,13,14,15);
}

// B fragment from an LDS-staged 64x32 tile (row stride 32 halves).
__device__ inline v16h load_frag_b_lds(const _Float16* tile, int col_local, int lane) {
  const int l = lane & 15;
  const int hi = (lane >= 16) ? 16 : 0;
  const _Float16* p = tile + (col_local + l) * 32 + hi;
  v8h lo = *(const v8h*)p;
  v8h hh = *(const v8h*)(p + 8);
  return __builtin_shufflevector(lo, hh, 0,1,2,3,4,5,6,7,8,9,10,11,12,13,14,15);
}

__device__ inline v8f wmma16(v16h a, v16h b, v8f c) {
  return __builtin_amdgcn_wmma_f32_16x16x32_f16(false, a, false, b,
                                                (short)0, c, false, false);
}

// ---------------------------------------------------------------------------
// Generic WMMA GEMM: C(MxN) = A(MxK,f16) @ Bt(NxK,f16)^T with fused epilogue.
// Block: 256 thr = 8 waves, tile 128x64; wave computes 32x32 (4 wmma accs).
// B tile (64x32 = 4KB) is double-buffered in LDS via async copies: every
// thread moves one b128; waits are per-wave ASYNCcnt + workgroup barrier.
// The last K-step is peeled so the steady-state loop is branch-free.
// mode 0: C = acc                (float out)
// mode 1: C += acc               (residual add, float out)
// mode 2: C16 = f16(acc)
// mode 3: C16 = f16(silu(acc) * aux16)   (SwiGLU gate path)
// ---------------------------------------------------------------------------
__global__ __launch_bounds__(256) void k_gemm(const _Float16* __restrict__ A,
                                              const _Float16* __restrict__ Bt,
                                              float* __restrict__ C,
                                              _Float16* __restrict__ C16,
                                              const _Float16* __restrict__ aux16,
                                              int N, int K, int mode) {
  __shared__ __align__(16) _Float16 btile[2][64 * 32];
  const int tid  = threadIdx.x;
  const int lane = tid & 31;
  const int wave = tid >> 5;
  const int row0   = blockIdx.x * 128 + (wave >> 1) * 32;
  const int colblk = blockIdx.y * 64;
  const int wcol   = (wave & 1) * 32;          // wave's column offset in tile
  const int KT = K >> 5;

  // Per-thread async-copy coordinates: 4 threads per tile row (64B/row).
  const int cb_n = tid >> 2;                    // tile row (B column) 0..63
  const int cb_k = (tid & 3) * 8;               // halfword offset in row
  const _Float16* gsrc = Bt + (size_t)(colblk + cb_n) * K + cb_k;
  const unsigned lds0 = (unsigned)(uintptr_t)&btile[0][tid * 8];
  const unsigned lds1 = (unsigned)(uintptr_t)&btile[1][tid * 8];

  // Preload tile 0.
  async_copy_b128(lds0, gsrc);

  v8f acc00 = {}, acc01 = {}, acc10 = {}, acc11 = {};
  // Steady state: branch-free (always a next tile to stage).
  for (int kt = 0; kt < KT - 1; ++kt) {
    const int k0  = kt << 5;
    const int cur = kt & 1;
    async_copy_b128(cur ? lds0 : lds1, gsrc + k0 + 32);  // stage tile kt+1
    wait_async_le1();     // own copies for tile kt complete
    __syncthreads();      // all waves' copies for tile kt visible

    __builtin_prefetch(A + (size_t)(row0 + (lane & 15)) * K + k0 + 32, 0, 1);
    v16h a0 = load_frag_a(A, K, row0,      k0, lane);
    v16h a1 = load_frag_a(A, K, row0 + 16, k0, lane);
    v16h b0 = load_frag_b_lds(&btile[cur][0], wcol,      lane);
    v16h b1 = load_frag_b_lds(&btile[cur][0], wcol + 16, lane);
    acc00 = wmma16(a0, b0, acc00);
    acc01 = wmma16(a0, b1, acc01);
    acc10 = wmma16(a1, b0, acc10);
    acc11 = wmma16(a1, b1, acc11);

    __syncthreads();      // btile[cur] is re-filled at step kt+1 (tile kt+2)
  }
  // Peeled last K-step.
  {
    const int kt  = KT - 1;
    const int k0  = kt << 5;
    const int cur = kt & 1;
    wait_async_0();
    __syncthreads();
    v16h a0 = load_frag_a(A, K, row0,      k0, lane);
    v16h a1 = load_frag_a(A, K, row0 + 16, k0, lane);
    v16h b0 = load_frag_b_lds(&btile[cur][0], wcol,      lane);
    v16h b1 = load_frag_b_lds(&btile[cur][0], wcol + 16, lane);
    acc00 = wmma16(a0, b0, acc00);
    acc01 = wmma16(a0, b1, acc01);
    acc10 = wmma16(a1, b0, acc10);
    acc11 = wmma16(a1, b1, acc11);
  }

  const int l = lane & 15;
  const int rofs = (lane >= 16) ? 8 : 0;
  const int col0 = colblk + wcol;
  v8f accs[2][2] = {{acc00, acc01}, {acc10, acc11}};
  for (int i = 0; i < 2; ++i)
    for (int j = 0; j < 2; ++j)
      for (int r = 0; r < 8; ++r) {
        const int row = row0 + i * 16 + rofs + r;
        const int col = col0 + j * 16 + l;
        const size_t idx = (size_t)row * N + col;
        const float v = accs[i][j][r];
        if (mode == 0)       C[idx] = v;
        else if (mode == 1)  C[idx] += v;
        else if (mode == 2)  C16[idx] = (_Float16)v;
        else {               // silu(acc) * up
          const float u = (float)aux16[idx];
          C16[idx] = (_Float16)((v / (1.f + __expf(-v))) * u);
        }
      }
}

// ---------------------------------------------------------------------------
// Input projection GEMM: x_tokens = video @ W_in (+bias / mask-token select),
// scattered into x at row (bt*608 + 32 + n). M=9216, K=768, N=512.
// (One-shot kernel; keeps the direct-global fragment path.)
// ---------------------------------------------------------------------------
__global__ __launch_bounds__(256) void k_prep_gemm(const _Float16* __restrict__ A,
                                                   const _Float16* __restrict__ Bt,
                                                   float* __restrict__ x,
                                                   const float* __restrict__ b_in,
                                                   const float* __restrict__ mask_token,
                                                   const float* __restrict__ mask_prob,
                                                   const float* __restrict__ mask_rand) {
  const int K = CIN, N = DM;
  const int lane = threadIdx.x & 31;
  const int wave = threadIdx.x >> 5;
  const int row0 = blockIdx.x * 128 + (wave >> 1) * 32;
  const int col0 = blockIdx.y * 64  + (wave & 1) * 32;

  v8f acc00 = {}, acc01 = {}, acc10 = {}, acc11 = {};
  for (int k0 = 0; k0 < K; k0 += 32) {
    v16h a0 = load_frag_a(A, K, row0,      k0, lane);
    v16h a1 = load_frag_a(A, K, row0 + 16, k0, lane);
    v16h b0 = load_frag_b(Bt, K, col0,      k0, lane);
    v16h b1 = load_frag_b(Bt, K, col0 + 16, k0, lane);
    acc00 = wmma16(a0, b0, acc00);
    acc01 = wmma16(a0, b1, acc01);
    acc10 = wmma16(a1, b0, acc10);
    acc11 = wmma16(a1, b1, acc11);
  }

  const int l = lane & 15;
  const int rofs = (lane >= 16) ? 8 : 0;
  v8f accs[2][2] = {{acc00, acc01}, {acc10, acc11}};
  for (int i = 0; i < 2; ++i)
    for (int j = 0; j < 2; ++j)
      for (int r = 0; r < 8; ++r) {
        const int row = row0 + i * 16 + rofs + r;   // token row in [0, 9216)
        const int col = col0 + j * 16 + l;
        const int bt = row / NTOK;
        const int n  = row - bt * NTOK;
        const bool masked = mask_rand[row] < mask_prob[bt];
        const float v = accs[i][j][r];
        x[(size_t)(bt * TOTS + NLAT + n) * DM + col] =
            masked ? mask_token[col] : (v + b_in[col]);
      }
}

// Broadcast latent tokens into the first 32 rows of each (b,t) chunk.
__global__ void k_latents(float* __restrict__ x, const float* __restrict__ lat) {
  const int tid = blockIdx.x * blockDim.x + threadIdx.x;
  if (tid >= BB * TT * NLAT * DM) return;
  const int d = tid & (DM - 1);
  const int l = (tid >> 9) & (NLAT - 1);
  const int bt = tid >> 14;
  x[(size_t)(bt * TOTS + l) * DM + d] = lat[l * DM + d];
}

// ---------------------------------------------------------------------------
// RMSNorm: one wave per row (wave32); output directly in f16 to feed WMMA.
// ---------------------------------------------------------------------------
__global__ __launch_bounds__(256) void k_rmsnorm(const float* __restrict__ x,
                                                 const float* __restrict__ g,
                                                 _Float16* __restrict__ out,
                                                 int M) {
  const int wave = threadIdx.x >> 5, lane = threadIdx.x & 31;
  const int row = blockIdx.x * 8 + wave;
  if (row >= M) return;
  const float* xr = x + (size_t)row * DM;
  float ss = 0.f;
  for (int d = lane; d < DM; d += 32) { const float t = xr[d]; ss += t * t; }
  for (int o = 16; o; o >>= 1) ss += __shfl_xor(ss, o, 32);
  const float r = rsqrtf(ss * (1.f / DM) + 1e-6f);
  _Float16* orow = out + (size_t)row * DM;
  for (int d = lane; d < DM; d += 32) orow[d] = (_Float16)(xr[d] * r * g[d]);
}

// ---------------------------------------------------------------------------
// RoPE applied in place to q (8 heads) and k (4 heads). Angles on the fly.
// Spatial: 2D rope by token (x,y); latents (pos<32) are identity.
// Temporal: 1D rope by t.
// ---------------------------------------------------------------------------
__global__ void k_rope(float* __restrict__ q, float* __restrict__ kk,
                       int M, int temporal) {
  const int tid = blockIdx.x * blockDim.x + threadIdx.x;
  if (tid >= M * 12 * 32) return;
  const int p   = tid & 31;
  const int hh  = (tid >> 5) % 12;
  const int row = tid / (12 * 32);

  float ang;
  if (temporal) {
    const int t = (row / TOTS) & (TT - 1);
    ang = (float)t * __powf(10000.f, -(float)p * (1.f / 32.f));
  } else {
    const int s = row % TOTS;
    if (s < NLAT) return;  // latent rows: cos=1, sin=0 -> identity
    const int tok = s - NLAT;
    const int xg = tok % 24, yg = tok / 24;
    ang = (p < 16) ? (float)xg * __powf(10000.f, -(float)p * (1.f / 16.f))
                   : (float)yg * __powf(10000.f, -(float)(p - 16) * (1.f / 16.f));
  }
  const float c = __cosf(ang), sn = __sinf(ang);
  float* buf;
  size_t base;
  if (hh < 8) { buf = q;  base = (size_t)row * (NH * HD)  + hh * HD + p; }
  else        { buf = kk; base = (size_t)row * (NKV * HD) + (hh - 8) * HD + p; }
  const float x1 = buf[base], x2 = buf[base + 32];
  buf[base]      = x1 * c - x2 * sn;
  buf[base + 32] = x2 * c + x1 * sn;
}

// ---------------------------------------------------------------------------
// Attention: one wave per (seq, head, query). Two-pass softmax with per-wave
// LDS score buffer; V accumulation coalesced across lanes. Output in f16.
// ---------------------------------------------------------------------------
__device__ inline int row_of(int seq, int pos, int temporal) {
  if (!temporal) return seq * TOTS + pos;
  const int b = seq / TOTS, tot = seq - b * TOTS;
  return (b * TT + pos) * TOTS + tot;
}

__global__ __launch_bounds__(256) void k_attn(const float* __restrict__ Q,
                                              const float* __restrict__ Kb,
                                              const float* __restrict__ Vb,
                                              _Float16* __restrict__ O,
                                              int S, int temporal) {
  __shared__ __align__(16) float sc[8][TOTS];
  __shared__ __align__(16) float qsh[8][HD];
  const int wave = threadIdx.x >> 5, lane = threadIdx.x & 31;
  const int gw = blockIdx.x * 8 + wave;
  const int qpos = gw % S;
  const int h    = (gw / S) & (NH - 1);
  const int seq  = gw / (S * NH);
  const int kvh  = h >> 1;

  const int rowq = row_of(seq, qpos, temporal);
  qsh[wave][lane]      = Q[(size_t)rowq * (NH * HD) + h * HD + lane];
  qsh[wave][lane + 32] = Q[(size_t)rowq * (NH * HD) + h * HD + lane + 32];

  // pass 1: scores + running max (lane-strided over keys)
  float mmax = -INFINITY;
  for (int j = lane; j < S; j += 32) {
    const bool valid = temporal ? (j <= qpos) : ((qpos < NLAT) || (j >= NLAT));
    float s = -INFINITY;
    if (valid) {
      const int rowj = row_of(seq, j, temporal);
      const float4* kp = (const float4*)(Kb + (size_t)rowj * (NKV * HD) + kvh * HD);
      const float4* qp = (const float4*)(&qsh[wave][0]);
      float a = 0.f;
#pragma unroll
      for (int d = 0; d < 16; ++d) {
        const float4 qa = qp[d], kb = kp[d];
        a += qa.x * kb.x + qa.y * kb.y + qa.z * kb.z + qa.w * kb.w;
      }
      s = a * 0.125f;   // 1/sqrt(64)
    }
    sc[wave][j] = s;
    mmax = fmaxf(mmax, s);
  }
  for (int o = 16; o; o >>= 1) mmax = fmaxf(mmax, __shfl_xor(mmax, o, 32));

  // pass 2: exp + sum
  float lsum = 0.f;
  for (int j = lane; j < S; j += 32) {
    const float p = __expf(sc[wave][j] - mmax);
    sc[wave][j] = p;
    lsum += p;
  }
  for (int o = 16; o; o >>= 1) lsum += __shfl_xor(lsum, o, 32);
  const float inv = 1.f / lsum;

  // pass 3: O = P @ V  (lanes own dims -> coalesced V reads)
  float a1 = 0.f, a2 = 0.f;
  for (int j = 0; j < S; ++j) {
    const float p = sc[wave][j];
    const int rowj = row_of(seq, j, temporal);
    const float* vp = Vb + (size_t)rowj * (NKV * HD) + kvh * HD;
    a1 += p * vp[lane];
    a2 += p * vp[lane + 32];
  }
  O[(size_t)rowq * (NH * HD) + h * HD + lane]      = (_Float16)(a1 * inv);
  O[(size_t)rowq * (NH * HD) + h * HD + lane + 32] = (_Float16)(a2 * inv);
}

// ---------------------------------------------------------------------------
// Final projection: latent = tanh(x[:, :, :32, :] @ W_out + b_out). Tiny.
// ---------------------------------------------------------------------------
__global__ void k_out(const float* __restrict__ x, const float* __restrict__ Wout,
                      const float* __restrict__ bout, float* __restrict__ out) {
  const int tid = blockIdx.x * blockDim.x + threadIdx.x;
  if (tid >= BB * TT * NLAT * 16) return;
  const int c  = tid & 15;
  const int l  = (tid >> 4) & (NLAT - 1);
  const int bt = tid >> 9;
  const float* xr = x + (size_t)(bt * TOTS + l) * DM;
  float acc = bout[c];
  for (int d = 0; d < DM; ++d) acc += xr[d] * Wout[d * 16 + c];
  out[tid] = tanhf(acc);
}

// ---------------------------------------------------------------------------
// f32 -> f16 conversion kernels (straight and K-major transpose for weights)
// ---------------------------------------------------------------------------
__global__ void k_cvt(const float* __restrict__ in, _Float16* __restrict__ out, int n) {
  const int i = blockIdx.x * blockDim.x + threadIdx.x;
  if (i < n) out[i] = (_Float16)in[i];
}
__global__ void k_cvtT(const float* __restrict__ W, _Float16* __restrict__ out,
                       int K, int N) {
  const int i = blockIdx.x * blockDim.x + threadIdx.x;
  if (i < K * N) {
    const int k = i / N, n = i - k * N;
    out[(size_t)n * K + k] = (_Float16)W[i];
  }
}

// ---------------------------------------------------------------------------
// Host launcher
// ---------------------------------------------------------------------------
extern "C" void kernel_launch(void* const* d_in, const int* in_sizes, int n_in,
                              void* d_out, int out_size, void* d_ws, size_t ws_size,
                              hipStream_t stream) {
  (void)in_sizes; (void)n_in; (void)out_size; (void)ws_size;

  const float* video       = (const float*)d_in[0];
  const float* mask_prob   = (const float*)d_in[1];
  const float* mask_rand   = (const float*)d_in[2];
  const float* W_in        = (const float*)d_in[3];
  const float* b_in        = (const float*)d_in[4];
  const float* mask_token  = (const float*)d_in[5];
  const float* latent_toks = (const float*)d_in[6];
  const float* norm1       = (const float*)d_in[7];
  const float* Wq          = (const float*)d_in[8];
  const float* Wk          = (const float*)d_in[9];
  const float* Wv          = (const float*)d_in[10];
  const float* Wo          = (const float*)d_in[11];
  const float* norm2       = (const float*)d_in[12];
  const float* Wg          = (const float*)d_in[13];
  const float* Wu          = (const float*)d_in[14];
  const float* Wd          = (const float*)d_in[15];
  const float* W_out       = (const float*)d_in[16];
  const float* b_out       = (const float*)d_in[17];
  float* out = (float*)d_out;

  char* ws = (char*)d_ws;
  size_t off = 0;
  auto alloc = [&](size_t bytes) {
    const size_t o = off;
    off += (bytes + 255) & ~(size_t)255;
    return o;
  };

  const int M = MROWS;  // 9728
  // Region A: u16 (M x 2048 f16); video16 (9216 x 768 f16) aliases its head
  // (video16 is dead before the first MLP writes u16).
  const size_t o_regA = alloc((size_t)M * MLPD * 2);
  _Float16* u16     = (_Float16*)(ws + o_regA);
  _Float16* video16 = u16;
  // Converted, K-major f16 weights
  const size_t o_wts = alloc((size_t)16121856 * 2);
  _Float16* wts = (_Float16*)(ws + o_wts);
  // Activations
  const size_t o_x   = alloc((size_t)M * DM * 4);
  const size_t o_h16 = alloc((size_t)M * DM * 2);
  // Region B: q|k|v (exactly M*1024 f32) reused later as act16 (M*2048 f16).
  const size_t o_regB = alloc((size_t)M * MLPD * 2);
  float*    q     = (float*)(ws + o_regB);
  float*    kbuf  = (float*)(ws + o_regB + (size_t)M * DM * 4);
  float*    vbuf  = (float*)(ws + o_regB + (size_t)M * DM * 4 + (size_t)M * NKV * HD * 4);
  _Float16* act16 = (_Float16*)(ws + o_regB);
  const size_t o_ao = alloc((size_t)M * DM * 2);
  _Float16* ao16 = (_Float16*)(ws + o_ao);

  float*    x      = (float*)(ws + o_x);
  _Float16* h16buf = (_Float16*)(ws + o_h16);

  // Weight table (all stored transposed: [N][K] f16)
  _Float16* winT = wts;
  size_t w = (size_t)CIN * DM;
  _Float16 *wqT[4], *wkT[4], *wvT[4], *woT[4], *wgT[4], *wuT[4], *wdT[4];
  for (int j = 0; j < 4; ++j) {
    wqT[j] = wts + w; w += (size_t)DM * DM;
    wkT[j] = wts + w; w += (size_t)DM * (NKV * HD);
    wvT[j] = wts + w; w += (size_t)DM * (NKV * HD);
    woT[j] = wts + w; w += (size_t)DM * DM;
    wgT[j] = wts + w; w += (size_t)DM * MLPD;
    wuT[j] = wts + w; w += (size_t)DM * MLPD;
    wdT[j] = wts + w; w += (size_t)MLPD * DM;
  }

  auto cvtT = [&](const float* Wp, _Float16* o, int K, int N) {
    const int n = K * N;
    k_cvtT<<<(n + 255) / 256, 256, 0, stream>>>(Wp, o, K, N);
  };
  // One-time f16 conversions (weights K-major, video straight)
  cvtT(W_in, winT, CIN, DM);
  for (int j = 0; j < 4; ++j) {
    cvtT(Wq + (size_t)j * DM * DM,          wqT[j], DM,   DM);
    cvtT(Wk + (size_t)j * DM * NKV * HD,    wkT[j], DM,   NKV * HD);
    cvtT(Wv + (size_t)j * DM * NKV * HD,    wvT[j], DM,   NKV * HD);
    cvtT(Wo + (size_t)j * DM * DM,          woT[j], DM,   DM);
    cvtT(Wg + (size_t)j * DM * MLPD,        wgT[j], DM,   MLPD);
    cvtT(Wu + (size_t)j * DM * MLPD,        wuT[j], DM,   MLPD);
    cvtT(Wd + (size_t)j * MLPD * DM,        wdT[j], MLPD, DM);
  }
  {
    const int n = MTOK * CIN;
    k_cvt<<<(n + 255) / 256, 256, 0, stream>>>(video, video16, n);
  }

  // Input projection + masking + latent concat
  {
    dim3 g(MTOK / 128, DM / 64);
    k_prep_gemm<<<g, 256, 0, stream>>>(video16, winT, x, b_in, mask_token,
                                       mask_prob, mask_rand);
    const int n = BB * TT * NLAT * DM;
    k_latents<<<(n + 255) / 256, 256, 0, stream>>>(x, latent_toks);
  }

  auto gemm = [&](const _Float16* A, const _Float16* Bt, float* C, _Float16* C16,
                  const _Float16* aux, int Mm, int N, int K, int mode) {
    dim3 g(Mm / 128, N / 64);
    k_gemm<<<g, 256, 0, stream>>>(A, Bt, C, C16, aux, N, K, mode);
  };

  for (int j = 0; j < 4; ++j) {
    const int temporal = (j == 3);  // layer i=4 -> temporal
    const int S = temporal ? TT : TOTS;
    const int nseq = temporal ? (BB * TOTS) : (BB * TT);
    const int nblocks = (nseq * NH * S) / 8;   // one wave per (seq, head, query)
    // attention
    k_rmsnorm<<<M / 8, 256, 0, stream>>>(x, norm1 + j * DM, h16buf, M);
    gemm(h16buf, wqT[j], q,    nullptr, nullptr, M, DM,       DM, 0);
    gemm(h16buf, wkT[j], kbuf, nullptr, nullptr, M, NKV * HD, DM, 0);
    gemm(h16buf, wvT[j], vbuf, nullptr, nullptr, M, NKV * HD, DM, 0);
    {
      const int n = M * 12 * 32;
      k_rope<<<(n + 255) / 256, 256, 0, stream>>>(q, kbuf, M, temporal);
    }
    k_attn<<<nblocks, 256, 0, stream>>>(q, kbuf, vbuf, ao16, S, temporal);
    gemm(ao16, woT[j], x, nullptr, nullptr, M, DM, DM, 1);  // x += attn @ Wo
    // MLP (SwiGLU)
    k_rmsnorm<<<M / 8, 256, 0, stream>>>(x, norm2 + j * DM, h16buf, M);
    gemm(h16buf, wuT[j], nullptr, u16,   nullptr, M, MLPD, DM, 2);
    gemm(h16buf, wgT[j], nullptr, act16, u16,     M, MLPD, DM, 3);
    gemm(act16, wdT[j], x, nullptr, nullptr, M, DM, MLPD, 1);  // x += act @ Wd
  }

  {
    const int n = BB * TT * NLAT * 16;
    k_out<<<(n + 255) / 256, 256, 0, stream>>>(x, W_out, b_out, out);
  }
}
